// ScaledDotProductAttention1_21371757265617
// MI455X (gfx1250) — matrix operand order repaired and checked by reading, not compiled, not run on hardware
//
#include <hip/hip_runtime.h>
#include <hip/hip_bf16.h>
#include <math.h>

// Problem constants
static constexpr int HNUM   = 16;
static constexpr int DK     = 64;
static constexpr int NSEQ   = 1024;
static constexpr int BS     = 4;
static constexpr int DMODEL = 1024;
static constexpr int TOPK   = 706;   // int(sigmoid(0.8) * 1024)

typedef __attribute__((ext_vector_type(16))) __bf16 v16bf;
typedef __attribute__((ext_vector_type(8)))  float  v8f;

union Frag { v16bf v; uint4 q[2]; unsigned int d[8]; unsigned short s[16]; };
union Acc  { v8f v; float f[8]; };

__device__ inline unsigned short f2bf(float f) {
  unsigned u = __float_as_uint(f);
  unsigned r = u + 0x7fffu + ((u >> 16) & 1u);   // round-to-nearest-even
  return (unsigned short)(r >> 16);
}
__device__ inline unsigned pkbf(float a, float b) {
  return (unsigned)f2bf(a) | ((unsigned)f2bf(b) << 16);
}

__device__ inline v8f wmma_bf16(const Frag& a, const Frag& b, v8f c) {
  return __builtin_amdgcn_wmma_f32_16x16x32_bf16(false, a.v, false, b.v,
                                                 (short)0, c, false, false);
}

// ---- CDNA5 async global->LDS copy (ASYNCcnt path), with safe fallback ------
#if __has_builtin(__builtin_amdgcn_global_load_async_to_lds_b128)
#define HAVE_ASYNC_LDS 1
typedef __attribute__((__vector_size__(4 * sizeof(int)))) int v4i_;
typedef __attribute__((address_space(1))) v4i_* gv4i_p;   // global (AS1)
typedef __attribute__((address_space(3))) v4i_* lv4i_p;   // LDS (AS3)
#endif

__device__ inline void cp16_g2l(const unsigned short* g, unsigned short* l) {
#ifdef HAVE_ASYNC_LDS
  __builtin_amdgcn_global_load_async_to_lds_b128(
      (gv4i_p)(void*)(const_cast<unsigned short*>(g)),
      (lv4i_p)(void*)l, 0, 0);
#else
  *(uint4*)l = *(const uint4*)g;
#endif
}
__device__ inline void wait_async_all() {
#ifdef HAVE_ASYNC_LDS
  asm volatile("s_wait_asynccnt 0" ::: "memory");
#endif
}

// ---- fragment loaders (ISA 7.12.2 layouts, wave32) -------------------------
// A (16x32, 16-bit): lanes 0-15 -> M=lane, K chunks {0..7, 16..23};
//                    lanes 16-31 -> M=lane-16, K chunks {8..15, 24..31}.
__device__ inline Frag load_a_bf16(const unsigned short* base, int ld, int lane) {
  Frag f; int m = lane & 15; int half = lane >> 4;
  const unsigned short* r = base + (size_t)m * ld;
  f.q[0] = *(const uint4*)(r + (half ? 8 : 0));
  f.q[1] = *(const uint4*)(r + (half ? 24 : 16));
  return f;
}
// B (32x16, 16-bit): lanes 0-15 -> N=lane, K chunks {0..7, 16..23};
//                    lanes 16-31 -> N=lane-16, K chunks {8..15, 24..31}.
__device__ inline Frag load_b_bf16(const unsigned short* base, int ld, int lane) {
  Frag f; int n = lane & 15; int half = lane >> 4;
#pragma unroll
  for (int s = 0; s < 8; ++s) {
    f.s[s]     = base[(size_t)(8 * half + s) * ld + n];
    f.s[8 + s] = base[(size_t)(16 + 8 * half + s) * ld + n];
  }
  return f;
}

// ---- wave32 reductions -----------------------------------------------------
__device__ inline int   wredsumi(int v)   { for (int m = 16; m >= 1; m >>= 1) v += __shfl_xor(v, m, 32); return v; }
__device__ inline float wredsumf(float v) { for (int m = 16; m >= 1; m >>= 1) v += __shfl_xor(v, m, 32); return v; }
__device__ inline float wredmaxf(float v) { for (int m = 16; m >= 1; m >>= 1) v = fmaxf(v, __shfl_xor(v, m, 32)); return v; }

// ---- fp32 -> bf16 bulk conversion (vectorized, 4 elems/thread) -------------
__global__ void f32_to_bf16_kernel(const float* __restrict__ x,
                                   unsigned short* __restrict__ y, int n4) {
  int i = blockIdx.x * blockDim.x + threadIdx.x;
  if (i >= n4) return;
  float4 f = ((const float4*)x)[i];
  uint2 o; o.x = pkbf(f.x, f.y); o.y = pkbf(f.z, f.w);
  ((uint2*)y)[i] = o;
}

// ---- sinusoidal positional encoding: pe[n,64] and its transpose ------------
__global__ void pe_kernel(unsigned short* pe, unsigned short* pet) {
  int idx = blockIdx.x * blockDim.x + threadIdx.x;   // NSEQ*32 threads
  if (idx >= NSEQ * 32) return;
  int pos = idx >> 5, i = idx & 31;
  float div = expf((float)(2 * i) * (-logf(10000.f) / (float)DK));
  float ang = (float)pos * div;
  float s = sinf(ang), c = cosf(ang);
  pe[pos * DK + 2 * i]     = f2bf(s);
  pe[pos * DK + 2 * i + 1] = f2bf(c);
  pet[(2 * i) * NSEQ + pos]     = f2bf(s);
  pet[(2 * i + 1) * NSEQ + pos] = f2bf(c);
}

// ---- transpose + convert weight [N,K] fp32 -> [K,N] bf16 -------------------
__global__ void transpose_bf16_kernel(const float* __restrict__ W,
                                      unsigned short* __restrict__ Wt) {
  int idx = blockIdx.x * blockDim.x + threadIdx.x;
  if (idx >= DMODEL * DMODEL) return;
  int n = idx / DMODEL, k = idx % DMODEL;
  Wt[(size_t)k * DMODEL + n] = f2bf(W[idx]);
}

// ---- generic WMMA GEMM: Y[M,N] = A[M,K] @ Wt[K,N] + bias[N] ----------------
// A is bf16 [M,1024]; the 16x1024 A strip of a block is contiguous (ld==K),
// so it is staged into LDS once via the async global->LDS path and shared by
// all 8 waves (each wave owns one 16x16 N tile).
// MODE 0: Q -> bf16 [b,h,n,DK] scaled 1/sqrt(DK); 1: K -> bf16 [b,h,DK,n];
// MODE 2: V -> bf16 [b,h,n,DV]; 3: O -> fp32 [M,N].
template <int MODE>
__global__ __launch_bounds__(256) void gemm_kernel(
    const unsigned short* __restrict__ A, const unsigned short* __restrict__ Wt,
    const float* __restrict__ bias,
    unsigned short* __restrict__ obf, float* __restrict__ of) {
  __shared__ unsigned short sA[16 * DMODEL];   // 32 KB A strip
  int t = threadIdx.x;
  int lane = t & 31, wave = t >> 5;
  int m0 = blockIdx.x * 16;
  int n0 = blockIdx.y * 128 + wave * 16;

  const unsigned short* astrip = A + (size_t)m0 * DMODEL;  // contiguous 32 KB
#pragma unroll
  for (int i = 0; i < 8; ++i) {
    int c = t + 256 * i;                      // 16-byte chunk index (2048 total)
    cp16_g2l(astrip + c * 8, &sA[c * 8]);
  }
  wait_async_all();
  __syncthreads();

  v8f acc = {};
  int m = lane & 15, half = lane >> 4;
  const unsigned short* arow = &sA[(size_t)m * DMODEL];
  for (int kb = 0; kb < DMODEL; kb += 32) {
    Frag a;
    a.q[0] = *(const uint4*)(arow + kb + (half ? 8 : 0));
    a.q[1] = *(const uint4*)(arow + kb + (half ? 24 : 16));
    Frag b = load_b_bf16(Wt + (size_t)kb * DMODEL + n0, DMODEL, lane);
    if (kb + 32 < DMODEL)
      __builtin_prefetch(Wt + (size_t)(kb + 32) * DMODEL + n0 + (size_t)lane * DMODEL, 0, 1);
    acc = wmma_bf16(a, b, acc);
  }

  Acc ac; ac.v = acc;
  int nl = lane & 15, col = n0 + nl;
  float bsv = bias[col];
#pragma unroll
  for (int r = 0; r < 8; ++r) {
    int mg = m0 + r + half * 8;
    float val = ac.f[r] + bsv;
    int b_ = mg >> 10, nrow = mg & 1023;
    int h = col >> 6, dd = col & 63;
    if constexpr (MODE == 0)
      obf[(((size_t)(b_ * HNUM + h)) * NSEQ + nrow) * DK + dd] = f2bf(val * 0.125f);
    else if constexpr (MODE == 1)
      obf[(((size_t)(b_ * HNUM + h)) * DK + dd) * NSEQ + nrow] = f2bf(val);
    else if constexpr (MODE == 2)
      obf[(((size_t)(b_ * HNUM + h)) * NSEQ + nrow) * DK + dd] = f2bf(val);
    else
      of[(size_t)mg * DMODEL + col] = val;
  }
}

// ---- fused attention: one block = one (b,h) x 16 query rows ----------------
__global__ __launch_bounds__(128) void attn_kernel(
    const unsigned short* __restrict__ q, const unsigned short* __restrict__ kt,
    const unsigned short* __restrict__ v, const unsigned short* __restrict__ pe,
    const unsigned short* __restrict__ pet, const float* __restrict__ biasp,
    unsigned short* __restrict__ ao) {
  __shared__ float satt[16 * NSEQ];            // 64 KB fp32 score tile
  __shared__ unsigned short sprob[16 * NSEQ];  // 32 KB bf16 prob tile (A-ready)
  __shared__ float srowscale[16];
  int lane = threadIdx.x & 31, wave = threadIdx.x >> 5;
  int bh = blockIdx.x >> 6, qt = blockIdx.x & 63;
  int qbase = qt * 16;
  int b_ = bh >> 4, h = bh & 15;

  // Phase 1: scores = (q/sqrt(dk)) @ k^T + pe @ pe^T, via WMMA
  const unsigned short* qp = q + (((size_t)bh) * NSEQ + qbase) * DK;
  Frag aq0 = load_a_bf16(qp, DK, lane);
  Frag aq1 = load_a_bf16(qp + 32, DK, lane);
  const unsigned short* pep = pe + (size_t)qbase * DK;
  Frag ap0 = load_a_bf16(pep, DK, lane);
  Frag ap1 = load_a_bf16(pep + 32, DK, lane);
  const unsigned short* ktp = kt + ((size_t)bh) * DK * NSEQ;
  for (int ct = wave; ct < 64; ct += 4) {
    int nb = ct * 16;
    Frag bk0 = load_b_bf16(ktp + nb, NSEQ, lane);
    Frag bk1 = load_b_bf16(ktp + (size_t)32 * NSEQ + nb, NSEQ, lane);
    Frag bp0 = load_b_bf16(pet + nb, NSEQ, lane);
    Frag bp1 = load_b_bf16(pet + (size_t)32 * NSEQ + nb, NSEQ, lane);
    v8f acc = {};
    acc = wmma_bf16(aq0, bk0, acc);
    acc = wmma_bf16(aq1, bk1, acc);
    acc = wmma_bf16(ap0, bp0, acc);
    acc = wmma_bf16(ap1, bp1, acc);
    Acc ac; ac.v = acc;
    int half = lane >> 4, nl = lane & 15;
#pragma unroll
    for (int r = 0; r < 8; ++r)
      satt[(r + half * 8) * NSEQ + nb + nl] = ac.f[r];
  }
  __syncthreads();

  // Phase 2: exact per-row top-k threshold (bitwise radix-select), masked exp;
  // probabilities written as bf16 straight into the WMMA-A-ready LDS tile.
  float bias = biasp[0];
  for (int row = wave; row < 16; row += 4) {
    float x[32]; unsigned key[32];
#pragma unroll
    for (int i = 0; i < 32; ++i) {
      float xv = satt[row * NSEQ + i * 32 + lane];
      x[i] = xv;
      unsigned u = __float_as_uint(xv);
      key[i] = (u & 0x80000000u) ? ~u : (u | 0x80000000u);  // order-preserving
    }
    float mx = -INFINITY;
#pragma unroll
    for (int i = 0; i < 32; ++i) mx = fmaxf(mx, x[i]);
    mx = wredmaxf(mx);
    unsigned prefix = 0; int need = TOPK;
    for (int bit = 31; bit >= 0; --bit) {
      unsigned hi = (bit == 31) ? 0u : (0xFFFFFFFFu << (bit + 1));
      unsigned tst = 1u << bit;
      int cnt = 0;
#pragma unroll
      for (int i = 0; i < 32; ++i)
        cnt += (int)((((key[i] & hi) == (prefix & hi)) && ((key[i] & tst) != 0u)) ? 1 : 0);
      cnt = wredsumi(cnt);
      if (cnt >= need) prefix |= tst; else need -= cnt;
    }
    unsigned thr = prefix;     // k-th largest key == threshold value
    float rs = 0.f;
#pragma unroll
    for (int i = 0; i < 32; ++i) {
      float e = (key[i] >= thr) ? __expf((x[i] - mx) * bias) : 0.f;
      rs += e;
      sprob[row * NSEQ + i * 32 + lane] = f2bf(e);
    }
    rs = wredsumf(rs);
    if (lane == 0) srowscale[row] = 1.f / rs;
  }
  __syncthreads();

  // Phase 3: out = P @ V via WMMA; A-frags are pure ds_load_b128 from sprob.
  {
    int dvb = wave * 16;
    const unsigned short* vp = v + ((size_t)bh) * NSEQ * DK + dvb;
    v8f acc = {};
    int m = lane & 15, half = lane >> 4;
    const unsigned short* prow = &sprob[(size_t)m * NSEQ];
    for (int kb = 0; kb < NSEQ; kb += 32) {
      Frag a;
      a.q[0] = *(const uint4*)(prow + kb + (half ? 8 : 0));
      a.q[1] = *(const uint4*)(prow + kb + (half ? 24 : 16));
      Frag b = load_b_bf16(vp + (size_t)kb * DK, DK, lane);
      acc = wmma_bf16(a, b, acc);
    }
    Acc ac; ac.v = acc;
    int nl = lane & 15;
#pragma unroll
    for (int r = 0; r < 8; ++r) {
      int mm = r + half * 8;
      float val = ac.f[r] * srowscale[mm];
      int nrow = qbase + mm;
      ao[((size_t)b_ * NSEQ + nrow) * DMODEL + h * DK + dvb + nl] = f2bf(val);
    }
  }
}

extern "C" void kernel_launch(void* const* d_in, const int* in_sizes, int n_in,
                              void* d_out, int out_size, void* d_ws, size_t ws_size,
                              hipStream_t stream) {
  (void)in_sizes; (void)n_in; (void)out_size; (void)ws_size;
  const float* queries = (const float*)d_in[0];
  const float* keys    = (const float*)d_in[1];
  const float* values  = (const float*)d_in[2];
  const float* Wq = (const float*)d_in[3];
  const float* bq = (const float*)d_in[4];
  const float* Wk = (const float*)d_in[5];
  const float* bk = (const float*)d_in[6];
  const float* Wv = (const float*)d_in[7];
  const float* bv = (const float*)d_in[8];
  const float* Wo = (const float*)d_in[9];
  const float* bo = (const float*)d_in[10];
  const float* biasp = (const float*)d_in[11];

  char* ws = (char*)d_ws;
  const size_t MB = 1024 * 1024;
  unsigned short* wtq = (unsigned short*)(ws + 0 * MB);
  unsigned short* wtk = (unsigned short*)(ws + 2 * MB);
  unsigned short* wtv = (unsigned short*)(ws + 4 * MB);
  unsigned short* wto = (unsigned short*)(ws + 6 * MB);
  unsigned short* qb  = (unsigned short*)(ws + 8 * MB);
  unsigned short* ktb = (unsigned short*)(ws + 16 * MB);
  unsigned short* vb  = (unsigned short*)(ws + 24 * MB);
  unsigned short* peb  = (unsigned short*)(ws + 32 * MB);
  unsigned short* petb = (unsigned short*)(ws + 32 * MB + 128 * 1024);
  unsigned short* xq = (unsigned short*)(ws + 33 * MB);  // reused as ao (bf16)
  unsigned short* xk = (unsigned short*)(ws + 41 * MB);
  unsigned short* xv = (unsigned short*)(ws + 49 * MB);
  unsigned short* aob = xq;   // xq consumed before attention writes ao
  float* out = (float*)d_out;

  pe_kernel<<<(NSEQ * 32 + 255) / 256, 256, 0, stream>>>(peb, petb);
  int tN = DMODEL * DMODEL;
  transpose_bf16_kernel<<<(tN + 255) / 256, 256, 0, stream>>>(Wq, wtq);
  transpose_bf16_kernel<<<(tN + 255) / 256, 256, 0, stream>>>(Wk, wtk);
  transpose_bf16_kernel<<<(tN + 255) / 256, 256, 0, stream>>>(Wv, wtv);
  transpose_bf16_kernel<<<(tN + 255) / 256, 256, 0, stream>>>(Wo, wto);

  int n4 = BS * NSEQ * DMODEL / 4;
  f32_to_bf16_kernel<<<(n4 + 255) / 256, 256, 0, stream>>>(queries, xq, n4);
  f32_to_bf16_kernel<<<(n4 + 255) / 256, 256, 0, stream>>>(keys,    xk, n4);
  f32_to_bf16_kernel<<<(n4 + 255) / 256, 256, 0, stream>>>(values,  xv, n4);

  dim3 gg(BS * NSEQ / 16, DMODEL / 128);   // (256, 8)
  gemm_kernel<0><<<gg, 256, 0, stream>>>(xq, wtq, bq, qb,  nullptr);
  gemm_kernel<1><<<gg, 256, 0, stream>>>(xk, wtk, bk, ktb, nullptr);
  gemm_kernel<2><<<gg, 256, 0, stream>>>(xv, wtv, bv, vb,  nullptr);

  attn_kernel<<<BS * HNUM * (NSEQ / 16), 128, 0, stream>>>(qb, ktb, vb, peb, petb, biasp, aob);

  gemm_kernel<3><<<gg, 256, 0, stream>>>(aob, wto, bo, nullptr, out);
}
